// CausalAttnBlockpp_67336497267017
// MI455X (gfx1250) — matrix operand order repaired and checked by reading, not compiled, not run on hardware
//
#include <hip/hip_runtime.h>

// ---------------------------------------------------------------------------
// CDNA5 (gfx1250) causal-attention block: GN -> QKV -> flash-attn -> proj+res
// Matmuls on v_wmma_f32_16x16x32_bf16 (bf16 in, f32 acc); K/V tiles staged in
// LDS by the Tensor Data Mover (tensor_load_to_lds, TENSORcnt), double-buffered.
// ---------------------------------------------------------------------------

typedef __attribute__((ext_vector_type(16))) __bf16 v16bf;
typedef __attribute__((ext_vector_type(8)))  float  v8f;
typedef __attribute__((ext_vector_type(4)))  unsigned int v4u;
typedef __attribute__((ext_vector_type(8)))  int    v8i32;
typedef __attribute__((ext_vector_type(4)))  int    v4i32;

#define CC   256          // channels / head dim
#define NN   4096         // tokens per batch (H*W)
#define WW   64           // width (causal axis)
#define SCALE 0.0625f     // 256^-0.5
#define NEG_BIG (-3.0e38f)

#define KROW 264          // K tile LDS row stride (256 + 8 elem TDM pad)
#define VROW 40           // V tile LDS row stride (32 + 8 elem TDM pad)

union Frag {
    v16bf v;
    uint4 q[2];
    unsigned short h[16];
};

static __device__ inline unsigned short f2bf(float f) {
    unsigned int u = __float_as_uint(f);
    u += 0x7FFFu + ((u >> 16) & 1u);          // round-to-nearest-even
    return (unsigned short)(u >> 16);
}

static __device__ inline v8f vzero8() {
    v8f z;
#pragma unroll
    for (int i = 0; i < 8; ++i) z[i] = 0.0f;
    return z;
}

// A-fragment (16x32 bf16, row-major source). rowbase = &A[row * ld], row = lane%16.
// ISA layout: element e -> K = e + (e>=8 ? 8 : 0) + (lane>=16 ? 8 : 0)
static __device__ inline Frag load_a_frag(const unsigned short* rowbase, int k0, int lane) {
    Frag f;
    int off = k0 + ((lane & 16) ? 8 : 0);
    f.q[0] = *(const uint4*)(rowbase + off);
    f.q[1] = *(const uint4*)(rowbase + off + 16);
    return f;
}

// B-fragment (32x16 bf16), source stored "B-transposed":
// rowbase = &Bt[col * ld], col = lane%16; element e -> K = e + (e>=8 ? 8 : 0) + (lane>=16 ? 8 : 0)
static __device__ inline Frag load_bt_frag(const unsigned short* rowbase, int k0, int lane) {
    Frag f;
    int off = k0 + ((lane & 16) ? 8 : 0);
    f.q[0] = *(const uint4*)(rowbase + off);
    f.q[1] = *(const uint4*)(rowbase + off + 16);
    return f;
}

static __device__ inline v8f wmma_bf16(const Frag& a, const Frag& b, v8f c) {
    return __builtin_amdgcn_wmma_f32_16x16x32_bf16(false, a.v, false, b.v,
                                                   (short)0, c, false, false);
}

// Workgroup-relative LDS byte offset (flat -> group addrspacecast).
static __device__ inline unsigned lds_off_of(const void* p) {
    return (unsigned)(unsigned long long)
        (const __attribute__((address_space(3))) void*)p;
}

// ---------------------------------------------------------------------------
// Tensor Data Mover: 2-D tile (tile0 x tile1 elems of 2 bytes) global -> LDS,
// with per-row LDS padding (pad_amt+1 DWORDs every 2^(pad_int+1) DWORDs).
// D# per ISA cdna5 ch.8: group0 {count=1, lds_addr, global_addr, type=2},
// group1 {data_size, dims, tile dims, stride, pad}; groups 2/3 zero (2-D).
// ---------------------------------------------------------------------------
static __device__ inline void tdm_load_2d(unsigned lds_off, unsigned long long gaddr,
                                          unsigned tile0, unsigned tile1,
                                          unsigned long long stride0,
                                          unsigned tdim0, unsigned tdim1,
                                          unsigned pad_int, unsigned pad_amt) {
    v4u g0;
    g0[0] = 1u;                                        // count=1 (valid user D#)
    g0[1] = lds_off;                                   // lds_addr [63:32]
    g0[2] = (unsigned)gaddr;                           // global_addr [95:64]
    g0[3] = (unsigned)(gaddr >> 32) | (2u << 30);      // ga hi + type=2 [127:96]

    unsigned w[8] = {0, 0, 0, 0, 0, 0, 0, 0};
    w[0]  = (1u << 16)                                 // data_size = 2 bytes
          | (1u << 20)                                 // pad_enable
          | (pad_int << 22) | (pad_amt << 25);
    w[1] |= (tdim0 & 0xffffu) << 16;                   // tensor_dim0 [79:48]
    w[2] |= tdim0 >> 16;
    w[2] |= (tdim1 & 0xffffu) << 16;                   // tensor_dim1 [111:80]
    w[3] |= tdim1 >> 16;
    w[3] |= (tile0 & 0xffffu) << 16;                   // tile_dim0 [127:112]
    w[4] |= tile1 & 0xffffu;                           // tile_dim1 [143:128]
    w[5]  = (unsigned)stride0;                         // tensor_dim0_stride [207:160]
    w[6]  = (unsigned)(stride0 >> 32) & 0xffffu;

    v8i32 g1;
#pragma unroll
    for (int i = 0; i < 8; ++i) g1[i] = (int)w[i];
    v4i32 z4; z4[0] = z4[1] = z4[2] = z4[3] = 0;
#if __clang_major__ >= 23
    v8i32 z8;
#pragma unroll
    for (int i = 0; i < 8; ++i) z8[i] = 0;
    __builtin_amdgcn_tensor_load_to_lds(g0, g1, z4, z4, z8, 0);
#else
    __builtin_amdgcn_tensor_load_to_lds(g0, g1, z4, z4, 0);
#endif
}

// ---------------------------------------------------------------------------
// K0: transpose + convert the four 256x256 f32 weights into bf16 wt[sel][d][c]
// ---------------------------------------------------------------------------
__global__ __launch_bounds__(256) void wt_kernel(const float* __restrict__ w0,
                                                 const float* __restrict__ w1,
                                                 const float* __restrict__ w2,
                                                 const float* __restrict__ w3,
                                                 unsigned short* __restrict__ wt) {
    int i = blockIdx.x * 256 + threadIdx.x;
    int sel = i >> 16;
    int rem = i & 65535;
    int d = rem >> 8, c = rem & 255;
    const float* w = (sel == 0) ? w0 : (sel == 1) ? w1 : (sel == 2) ? w2 : w3;
    wt[i] = f2bf(w[c * CC + d]);
}

// ---------------------------------------------------------------------------
// K1: per-pixel GroupNorm (32 groups x 8 ch). One lane = one (pixel, group).
// ---------------------------------------------------------------------------
__global__ __launch_bounds__(256) void gn_kernel(const float* __restrict__ x,
                                                 const float* __restrict__ gw,
                                                 const float* __restrict__ gb,
                                                 unsigned short* __restrict__ hn) {
    int t  = blockIdx.x * 256 + threadIdx.x;
    int g  = t & 31;
    int pg = t >> 5;
    int b  = pg >> 12;
    int p  = pg & 4095;
    const float* xp = x + (size_t)b * CC * NN + p;
    int c0 = g * 8;
    float v[8], s = 0.f, ss = 0.f;
#pragma unroll
    for (int i = 0; i < 8; ++i) {
        float val = xp[(size_t)(c0 + i) * NN];
        v[i] = val; s += val; ss += val * val;
    }
    float mean = s * 0.125f;
    float var  = ss * 0.125f - mean * mean;
    var = var < 0.f ? 0.f : var;
    float inv = rsqrtf(var + 1e-6f);
    unsigned short* out = hn + (size_t)pg * CC + c0;
#pragma unroll
    for (int i = 0; i < 8; ++i)
        out[i] = f2bf((v[i] - mean) * inv * gw[c0 + i] + gb[c0 + i]);
}

// ---------------------------------------------------------------------------
// K2: Y[m,d] = sum_c A[m,c]*W[c,d] + bias[d]; 8 waves, block tile 128x64.
// transposed=0 -> Y[m*256+d]; transposed=1 -> Y[d*M+m]  (V^T layout)
// ---------------------------------------------------------------------------
__global__ __launch_bounds__(256) void gemm_bf16(const unsigned short* __restrict__ A,
                                                 const unsigned short* __restrict__ Bt,
                                                 const float* __restrict__ bias,
                                                 unsigned short* __restrict__ Y,
                                                 int M, int transposed) {
    int lane = threadIdx.x & 31, wave = threadIdx.x >> 5;
    int wm = wave >> 1, wn = wave & 1;
    int m0 = blockIdx.x * 128 + wm * 32;
    int d0 = blockIdx.y * 64 + wn * 32;
    int l15 = lane & 15, half = lane >> 4;

    const unsigned short* arow[2];
    const unsigned short* brow[2];
#pragma unroll
    for (int i = 0; i < 2; ++i) arow[i] = A  + (size_t)(m0 + i * 16 + l15) * CC;
#pragma unroll
    for (int j = 0; j < 2; ++j) brow[j] = Bt + (size_t)(d0 + j * 16 + l15) * CC;

    v8f acc[2][2];
#pragma unroll
    for (int i = 0; i < 2; ++i)
#pragma unroll
        for (int j = 0; j < 2; ++j) acc[i][j] = vzero8();

#pragma unroll
    for (int kc = 0; kc < CC; kc += 32) {
        Frag a[2], bf[2];
#pragma unroll
        for (int i = 0; i < 2; ++i) a[i]  = load_a_frag(arow[i], kc, lane);
#pragma unroll
        for (int j = 0; j < 2; ++j) bf[j] = load_bt_frag(brow[j], kc, lane);
#pragma unroll
        for (int i = 0; i < 2; ++i)
#pragma unroll
            for (int j = 0; j < 2; ++j) acc[i][j] = wmma_bf16(a[i], bf[j], acc[i][j]);
    }

#pragma unroll
    for (int j = 0; j < 2; ++j) {
        int d = d0 + j * 16 + l15;
        float bj = bias[d];
#pragma unroll
        for (int i = 0; i < 2; ++i)
#pragma unroll
            for (int r = 0; r < 8; ++r) {
                int m = m0 + i * 16 + r + 8 * half;
                float val = acc[i][j][r] + bj;
                if (!transposed) Y[(size_t)m * CC + d] = f2bf(val);
                else             Y[(size_t)d * M + m] = f2bf(val);
            }
    }
}

// ---------------------------------------------------------------------------
// K3: flash attention, TDM-staged K/V tiles (double buffered in LDS).
// 4 waves/block, wave = 16 query rows; key tiles of 32; mask (key%64)<=(q%64).
// Q,K bf16 [b*4096+n, c];  VT bf16 [c, b*4096+n];  H bf16 out [b*4096+n, c].
// ---------------------------------------------------------------------------
__global__ __launch_bounds__(128) void attn_kernel(const unsigned short* __restrict__ Qg,
                                                   const unsigned short* __restrict__ Kg,
                                                   const unsigned short* __restrict__ VTg,
                                                   unsigned short* __restrict__ Hg) {
    __shared__ __align__(16) unsigned short ktile[2][32 * KROW];    // 2 x 16.5KB
    __shared__ __align__(16) unsigned short vtile[2][256 * VROW];   // 2 x 20KB
    __shared__ __align__(16) unsigned short plds[4][16 * 32];       // P transpose

    int lane = threadIdx.x & 31, wave = threadIdx.x >> 5;
    int l15 = lane & 15, half = lane >> 4;
    int b  = blockIdx.y;
    int q0 = blockIdx.x * 64 + wave * 16;

    const unsigned short* Q  = Qg  + (size_t)b * NN * CC;
    const unsigned short* K  = Kg  + (size_t)b * NN * CC;
    const unsigned short* VT = VTg + (size_t)b * NN;     // per-c stride = 4*NN
    unsigned short*       H  = Hg  + (size_t)b * NN * CC;

    // Preload this wave's Q fragments (16 rows x 256, 8 K-chunks).
    Frag qf[8];
    const unsigned short* qrow = Q + (size_t)(q0 + l15) * CC;
#pragma unroll
    for (int kc = 0; kc < 8; ++kc) qf[kc] = load_a_frag(qrow, kc * 32, lane);

    v8f o[16];
#pragma unroll
    for (int cg = 0; cg < 16; ++cg) o[cg] = vzero8();
    float mrow[8], lrow[8];
    int   qw[8];
#pragma unroll
    for (int r = 0; r < 8; ++r) {
        mrow[r] = NEG_BIG; lrow[r] = 0.0f;
        qw[r] = (q0 + r + 8 * half) & (WW - 1);
    }

    // TDM issue for key tile at kbase into buffer buf (wave 0 only).
    // K tile: 32 rows x 256 elems, row pad 16B  -> ktile rows of KROW elems.
    // V tile: 256 rows x 32 elems, row pad 16B  -> vtile rows of VROW elems.
    auto issue_tiles = [&](int kbase, int buf) {
        tdm_load_2d(lds_off_of(&ktile[buf][0]),
                    (unsigned long long)(const void*)(K + (size_t)kbase * CC),
                    /*tile0=*/CC, /*tile1=*/32, /*stride0=*/CC,
                    /*tdim0=*/CC, /*tdim1=*/NN,
                    /*pad_int=*/6, /*pad_amt=*/3);      // pad 4 DW / 128 DW
        tdm_load_2d(lds_off_of(&vtile[buf][0]),
                    (unsigned long long)(const void*)(VT + kbase),
                    /*tile0=*/32, /*tile1=*/CC, /*stride0=*/4 * NN,
                    /*tdim0=*/32, /*tdim1=*/CC,
                    /*pad_int=*/3, /*pad_amt=*/3);      // pad 4 DW / 16 DW
    };

    if (wave == 0) {
        issue_tiles(0, 0);
        __builtin_amdgcn_s_wait_tensorcnt((short)0);
    }
    __syncthreads();

    for (int kb = 0; kb < NN; kb += 32) {
        int cur = (kb >> 5) & 1;
        if (wave == 0 && kb + 32 < NN) issue_tiles(kb + 32, cur ^ 1);

        const unsigned short* kt = &ktile[cur][0];
        const unsigned short* vt = &vtile[cur][0];

        // ---- S = Q K^T over the 32-key tile (two 16-col subtiles) ----
        const unsigned short* kr0 = kt + (size_t)l15 * KROW;
        const unsigned short* kr1 = kt + (size_t)(16 + l15) * KROW;
        v8f s0 = vzero8(), s1 = vzero8();
#pragma unroll
        for (int kc = 0; kc < 8; ++kc) {
            Frag b0 = load_bt_frag(kr0, kc * 32, lane);
            s0 = wmma_bf16(qf[kc], b0, s0);
            Frag b1 = load_bt_frag(kr1, kc * 32, lane);
            s1 = wmma_bf16(qf[kc], b1, s1);
        }

        // ---- scale + causal-in-W mask ----
        int kw0 = (kb + l15) & (WW - 1);
        int kw1 = (kb + 16 + l15) & (WW - 1);
        float p0[8], p1[8], alpha[8];
#pragma unroll
        for (int r = 0; r < 8; ++r) {
            p0[r] = (kw0 <= qw[r]) ? s0[r] * SCALE : NEG_BIG;
            p1[r] = (kw1 <= qw[r]) ? s1[r] * SCALE : NEG_BIG;
        }

        // ---- online softmax stats (reduce across the 16-lane half per row) ----
#pragma unroll
        for (int r = 0; r < 8; ++r) {
            float rm = fmaxf(p0[r], p1[r]);
            rm = fmaxf(rm, __shfl_xor(rm, 1, 32));
            rm = fmaxf(rm, __shfl_xor(rm, 2, 32));
            rm = fmaxf(rm, __shfl_xor(rm, 4, 32));
            rm = fmaxf(rm, __shfl_xor(rm, 8, 32));
            float mn = fmaxf(mrow[r], rm);
            alpha[r] = __expf(mrow[r] - mn);
            mrow[r]  = mn;
            p0[r] = __expf(p0[r] - mn);
            p1[r] = __expf(p1[r] - mn);
            float rs = p0[r] + p1[r];
            rs += __shfl_xor(rs, 1, 32);
            rs += __shfl_xor(rs, 2, 32);
            rs += __shfl_xor(rs, 4, 32);
            rs += __shfl_xor(rs, 8, 32);
            lrow[r] = lrow[r] * alpha[r] + rs;
        }

        // ---- rescale accumulator ----
#pragma unroll
        for (int cg = 0; cg < 16; ++cg)
#pragma unroll
            for (int r = 0; r < 8; ++r) o[cg][r] *= alpha[r];

        // ---- D-layout -> A-layout transpose of P via per-wave LDS slice ----
        unsigned short* pw = &plds[wave][0];
#pragma unroll
        for (int r = 0; r < 8; ++r) {
            int row = r + 8 * half;
            pw[row * 32 + l15]      = f2bf(p0[r]);
            pw[row * 32 + 16 + l15] = f2bf(p1[r]);
        }
        Frag pa = load_a_frag(&plds[wave][l15 * 32], 0, lane);

        // ---- O += P V (B-frags from TDM-staged V^T tile) ----
#pragma unroll
        for (int cg = 0; cg < 16; ++cg) {
            const unsigned short* vr = vt + (size_t)(cg * 16 + l15) * VROW;
            Frag vf = load_bt_frag(vr, 0, lane);
            o[cg] = wmma_bf16(pa, vf, o[cg]);
        }

        // publish tile kb+32 / WAR-guard buffer reuse
        if (wave == 0) __builtin_amdgcn_s_wait_tensorcnt((short)0);
        __syncthreads();
    }

    // ---- finalize: O / l, store bf16 row-major ----
#pragma unroll
    for (int r = 0; r < 8; ++r) lrow[r] = 1.0f / lrow[r];
#pragma unroll
    for (int cg = 0; cg < 16; ++cg) {
        int d = cg * 16 + l15;
#pragma unroll
        for (int r = 0; r < 8; ++r) {
            int m = q0 + r + 8 * half;
            H[(size_t)m * CC + d] = f2bf(o[cg][r] * lrow[r]);
        }
    }
}

// ---------------------------------------------------------------------------
// K4: out[b,d,n] = x[b,d,n] + sum_c h[m,c]*w3[c,d] + b3[d]   (f32 output)
// ---------------------------------------------------------------------------
__global__ __launch_bounds__(256) void gemm_out(const unsigned short* __restrict__ A,
                                                const unsigned short* __restrict__ Bt,
                                                const float* __restrict__ bias,
                                                const float* __restrict__ x,
                                                float* __restrict__ out) {
    int lane = threadIdx.x & 31, wave = threadIdx.x >> 5;
    int wm = wave >> 1, wn = wave & 1;
    int m0 = blockIdx.x * 128 + wm * 32;
    int d0 = blockIdx.y * 64 + wn * 32;
    int l15 = lane & 15, half = lane >> 4;

    const unsigned short* arow[2];
    const unsigned short* brow[2];
#pragma unroll
    for (int i = 0; i < 2; ++i) arow[i] = A  + (size_t)(m0 + i * 16 + l15) * CC;
#pragma unroll
    for (int j = 0; j < 2; ++j) brow[j] = Bt + (size_t)(d0 + j * 16 + l15) * CC;

    v8f acc[2][2];
#pragma unroll
    for (int i = 0; i < 2; ++i)
#pragma unroll
        for (int j = 0; j < 2; ++j) acc[i][j] = vzero8();

#pragma unroll
    for (int kc = 0; kc < CC; kc += 32) {
        Frag a[2], bf[2];
#pragma unroll
        for (int i = 0; i < 2; ++i) a[i]  = load_a_frag(arow[i], kc, lane);
#pragma unroll
        for (int j = 0; j < 2; ++j) bf[j] = load_bt_frag(brow[j], kc, lane);
#pragma unroll
        for (int i = 0; i < 2; ++i)
#pragma unroll
            for (int j = 0; j < 2; ++j) acc[i][j] = wmma_bf16(a[i], bf[j], acc[i][j]);
    }

#pragma unroll
    for (int j = 0; j < 2; ++j) {
        int d = d0 + j * 16 + l15;
        float bj = bias[d];
#pragma unroll
        for (int i = 0; i < 2; ++i)
#pragma unroll
            for (int r = 0; r < 8; ++r) {
                int m = m0 + i * 16 + r + 8 * half;
                int bb = m >> 12, n = m & 4095;
                size_t idx = ((size_t)(bb * CC + d) << 12) + n;
                out[idx] = x[idx] + acc[i][j][r] + bj;
            }
    }
}

// ---------------------------------------------------------------------------
extern "C" void kernel_launch(void* const* d_in, const int* in_sizes, int n_in,
                              void* d_out, int out_size, void* d_ws, size_t ws_size,
                              hipStream_t stream) {
    const float* x  = (const float*)d_in[0];
    const float* gw = (const float*)d_in[1];
    const float* gb = (const float*)d_in[2];
    const float* w0 = (const float*)d_in[3];
    const float* b0 = (const float*)d_in[4];
    const float* w1 = (const float*)d_in[5];
    const float* b1 = (const float*)d_in[6];
    const float* w2 = (const float*)d_in[7];
    const float* b2 = (const float*)d_in[8];
    const float* w3 = (const float*)d_in[9];
    const float* b3 = (const float*)d_in[10];
    float* out = (float*)d_out;

    const size_t M = 4 * (size_t)NN;                 // 16384 rows total
    char* ws = (char*)d_ws;
    unsigned short* wt = (unsigned short*)ws;                 // 4*65536 bf16
    unsigned short* hn = (unsigned short*)(ws + (4u << 17));  // +512KB
    unsigned short* qb = hn + M * CC;
    unsigned short* kb = qb + M * CC;
    unsigned short* vt = kb + M * CC;                // stored [c][b*4096+n]
    unsigned short* hb = vt + M * CC;

    wt_kernel<<<1024, 256, 0, stream>>>(w0, w1, w2, w3, wt);
    gn_kernel<<<2048, 256, 0, stream>>>(x, gw, gb, hn);
    gemm_bf16<<<dim3(128, 4), 256, 0, stream>>>(hn, wt,             b0, qb, (int)M, 0);
    gemm_bf16<<<dim3(128, 4), 256, 0, stream>>>(hn, wt + 65536,     b1, kb, (int)M, 0);
    gemm_bf16<<<dim3(128, 4), 256, 0, stream>>>(hn, wt + 2 * 65536, b2, vt, (int)M, 1);
    attn_kernel<<<dim3(64, 4), 128, 0, stream>>>(qb, kb, vt, hb);
    gemm_out<<<dim3(128, 4), 256, 0, stream>>>(hb, wt + 3 * 65536, b3, x, out);
}
